// ATA_37099927503045
// MI455X (gfx1250) — matrix operand order repaired and checked by reading, not compiled, not run on hardware
//
#include <hip/hip_runtime.h>
#include <hip/hip_bf16.h>

// ---------------------------------------------------------------------------
// MI455X (gfx1250) fused sparse-attention pipeline.
//   pool_ctx   : ctx -> ctxpool (B,384,49)  f32      (pool commutes with 1x1 conv)
//   pool_v     : x   -> vbuf    (B,G,d=64,k2=256) bf16   (stored as B^T for WMMA)
//   kproj      : k = wk * ctxpool -> kbuf (B,G,s=64(pad),d=64) bf16 (B^T layout)
//   qproj      : q = wq * x -> qbuf (B,G,n,64) bf16      (128x128 WMMA GEMM tiles)
//   attn_fused : per 128-pixel tile: q.kT -> topk32 -> omega GEMM -> softmax
//                -> omega.v, all resident in 256KB LDS; wd DMA'd in via TDM.
//   proj       : y = proj_w * out + bias + x  (WMMA GEMM, f32 epilogue)
// All matrix math: v_wmma_f32_16x16x32_bf16, f32 accumulate.
// ---------------------------------------------------------------------------

typedef __bf16 bf16_t;
typedef __attribute__((ext_vector_type(16))) __bf16 v16bf;
typedef __attribute__((ext_vector_type(8)))  __bf16 v8bf;
typedef __attribute__((ext_vector_type(8)))  float  v8f;
typedef __attribute__((ext_vector_type(4)))  unsigned int u32x4;
typedef __attribute__((ext_vector_type(8)))  int i32x8;
typedef __attribute__((ext_vector_type(4)))  int i32x4;

#define HW   12544
#define CIN  768
#define CC   384
#define NG   12
#define HD   64
#define S2   49
#define K2   256
#define TOPK 32

__device__ __forceinline__ v8f zero_v8f() {
  v8f z;
#pragma unroll
  for (int i = 0; i < 8; ++i) z[i] = 0.0f;
  return z;
}

// A-fragment (16x32 bf16, M rows x K): lane l<16 -> row l, K pairs at
// {kb..kb+7} and {kb+16..kb+23}, kb = (l>>4)*8.  (ISA 7.12.2)
__device__ __forceinline__ v16bf load_a_frag(const bf16_t* base, int pitch, int lane) {
  const int r  = lane & 15;
  const int kb = (lane >> 4) * 8;
  const bf16_t* p = base + r * pitch + kb;
  union { v16bf v; v8bf h[2]; } u;
  u.h[0] = *(const v8bf*)p;
  u.h[1] = *(const v8bf*)(p + 16);
  return u.v;
}

// B-fragment (32x16 bf16, K x N cols): lane l holds column n=l&15, K at
// {kb..kb+7} and {kb+16..kb+23}, kb = (l>>4)*8.  Source is B^T stored row-major [n][k] in LDS.
__device__ __forceinline__ v16bf load_b_frag(const bf16_t* baseT, int pitch, int lane) {
  const int n  = lane & 15;
  const int kb = (lane >> 4) * 8;
  const bf16_t* p = baseT + n * pitch + kb;
  union { v16bf v; v8bf h[2]; } u;
  u.h[0] = *(const v8bf*)p;
  u.h[1] = *(const v8bf*)(p + 16);
  return u.v;
}

#define WMMA_BF16(ACC, A, B) \
  (ACC) = __builtin_amdgcn_wmma_f32_16x16x32_bf16(false, (A), false, (B), (short)0, (ACC), false, false)

// ---------------------------------------------------------------------------
// Pool kernels (memory bound; ~115MB reads total)
// ---------------------------------------------------------------------------
__global__ __launch_bounds__(256) void pool_ctx_kernel(const float* __restrict__ ctx,
                                                       float* __restrict__ ctxpool) {
  int idx = blockIdx.x * 256 + threadIdx.x;           // [b][c][s], 2*384*49
  if (idx >= 2 * CC * S2) return;
  int s = idx % S2, c = (idx / S2) % CC, b = idx / (CC * S2);
  int i = s / 7, j = s % 7;
  const float* p = ctx + ((size_t)(b * CC + c) * 112 + i * 16) * 112 + j * 16;
  float acc = 0.f;
#pragma unroll 4
  for (int r = 0; r < 16; ++r)
#pragma unroll
    for (int cc2 = 0; cc2 < 16; ++cc2) acc += p[r * 112 + cc2];
  ctxpool[idx] = acc * (1.0f / 256.0f);
}

__global__ __launch_bounds__(256) void pool_v_kernel(const float* __restrict__ x,
                                                     bf16_t* __restrict__ vbuf) {
  int idx = blockIdx.x * 256 + threadIdx.x;           // [b][c][k2], 2*768*256
  int k2 = idx & 255, c = (idx >> 8) % CIN, b = idx / (CIN * K2);
  int i = k2 >> 4, j = k2 & 15;
  const float* p = x + ((size_t)(b * CIN + c) * 112 + i * 7) * 112 + j * 7;
  float acc = 0.f;
#pragma unroll
  for (int r = 0; r < 7; ++r)
#pragma unroll
    for (int cc2 = 0; cc2 < 7; ++cc2) acc += p[r * 112 + cc2];
  vbuf[idx] = (bf16_t)(acc * (1.0f / 49.0f));         // layout [b][g][d][k2] == [b][c][k2]
}

// ---------------------------------------------------------------------------
// kproj: k = wk(768x384) * ctxpool(384x49) -> kbuf [b][g][s(64 pad)][d] bf16
// ---------------------------------------------------------------------------
__global__ __launch_bounds__(256) void kproj_kernel(const float* __restrict__ wk,
                                                    const float* __restrict__ ctxpool,
                                                    bf16_t* __restrict__ kbuf) {
  const int mBase = blockIdx.x * 128;
  const int b = blockIdx.y;
  __shared__ bf16_t sA[128 * 40];   // [m][k], pitch 40
  __shared__ bf16_t sB[64 * 40];    // B^T [s][k], pitch 40
  const int t = threadIdx.x, lane = t & 31, w = t >> 5;
  v8f acc[4];
#pragma unroll
  for (int i = 0; i < 4; ++i) acc[i] = zero_v8f();

  for (int k0 = 0; k0 < CC; k0 += 32) {
    { // A tile: wk rows
      int m = t >> 1, c0 = (t & 1) * 16;
      const float* src = wk + (size_t)(mBase + m) * CC + k0 + c0;
      bf16_t* dst = sA + m * 40 + c0;
#pragma unroll
      for (int j = 0; j < 16; ++j) dst[j] = (bf16_t)src[j];
    }
    { // B^T tile: sB[s][kk] = ctxpool[b][k0+kk][s], zero-pad s>=49
      int s = t >> 2, kk0 = (t & 3) * 8;
#pragma unroll
      for (int j = 0; j < 8; ++j) {
        float v = (s < S2) ? ctxpool[((size_t)b * CC + k0 + kk0 + j) * S2 + s] : 0.0f;
        sB[s * 40 + kk0 + j] = (bf16_t)v;
      }
    }
    __syncthreads();
    v16bf af = load_a_frag(sA + (w * 16) * 40, 40, lane);
#pragma unroll
    for (int nt = 0; nt < 4; ++nt) {
      v16bf bfr = load_b_frag(sB + (nt * 16) * 40, 40, lane);
      WMMA_BF16(acc[nt], af, bfr);
    }
    __syncthreads();
  }
  const int rowAdd = (lane >> 4) * 8, col = lane & 15;
#pragma unroll
  for (int nt = 0; nt < 4; ++nt)
#pragma unroll
    for (int r = 0; r < 8; ++r) {
      int o = mBase + w * 16 + r + rowAdd;
      int s = nt * 16 + col;
      int g = o >> 6, d = o & 63;
      kbuf[(((size_t)b * NG + g) * 64 + s) * 64 + d] = (bf16_t)acc[nt][r];
    }
}

// ---------------------------------------------------------------------------
// qproj: q = wq(768x768) * x(768xHW) -> qbuf [b][g][n][d] bf16. 128x128 tiles.
// ---------------------------------------------------------------------------
__global__ __launch_bounds__(256) void qproj_kernel(const float* __restrict__ x,
                                                    const float* __restrict__ wq,
                                                    bf16_t* __restrict__ qbuf) {
  const int nBase = blockIdx.x * 128;
  const int mBase = blockIdx.y * 128;
  const int b = blockIdx.z;
  __shared__ bf16_t sA[128 * 40];   // wq tile [m][k]
  __shared__ bf16_t sB[128 * 40];   // x  tile B^T [n][k]
  const int t = threadIdx.x, lane = t & 31, w = t >> 5;
  const float* xb = x + (size_t)b * CIN * HW;
  v8f acc[8];
#pragma unroll
  for (int i = 0; i < 8; ++i) acc[i] = zero_v8f();

  for (int k0 = 0; k0 < CIN; k0 += 32) {
    { // A: wq[mBase+m][k0..k0+31]
      int m = t >> 1, c0 = (t & 1) * 16;
      const float* src = wq + (size_t)(mBase + m) * CIN + k0 + c0;
      bf16_t* dst = sA + m * 40 + c0;
#pragma unroll
      for (int j = 0; j < 16; ++j) dst[j] = (bf16_t)src[j];
    }
    { // B^T: sB[n][kk] = x[b][k0+kk][nBase+n]  (coalesced read along n)
      int kk = t >> 3, n0 = (t & 7) * 16;
      const float* src = xb + (size_t)(k0 + kk) * HW + nBase + n0;
#pragma unroll
      for (int j = 0; j < 16; ++j) sB[(n0 + j) * 40 + kk] = (bf16_t)src[j];
    }
    __syncthreads();
    if (k0 + 32 < CIN) {  // gfx1250 global_prefetch_b8 for next tiles
      __builtin_prefetch(wq + (size_t)(mBase + (t >> 1)) * CIN + k0 + 32, 0, 1);
      __builtin_prefetch(xb + (size_t)(k0 + 32 + (t >> 3)) * HW + nBase + (t & 7) * 16, 0, 1);
    }
    v16bf af = load_a_frag(sA + (w * 16) * 40, 40, lane);
#pragma unroll
    for (int nt = 0; nt < 8; ++nt) {
      v16bf bfr = load_b_frag(sB + (nt * 16) * 40, 40, lane);
      WMMA_BF16(acc[nt], af, bfr);
    }
    __syncthreads();
  }
  const int rowAdd = (lane >> 4) * 8, col = lane & 15;
#pragma unroll
  for (int nt = 0; nt < 8; ++nt)
#pragma unroll
    for (int r = 0; r < 8; ++r) {
      int o = mBase + w * 16 + r + rowAdd;
      int n = nBase + nt * 16 + col;
      int g = o >> 6, d = o & 63;
      qbuf[(((size_t)b * NG + g) * HW + n) * 64 + d] = (bf16_t)acc[nt][r];
    }
}

// ---------------------------------------------------------------------------
// Fused attention: per (b,g, 128-pixel tile).
// LDS map (bytes, 256KB total, phase-reused):
//   sWd  @0      [256][72] bf16  36864   (B^T of wd, k-dim padded 49->64)
//   sKT  @36864  [64][72]  bf16   9216   (B^T of k: [s][d])
//   sQ   @46080  [128][72] bf16  18432   (A: q tile; later reused for sparse A)
//   sAt  @64512  [128][64] f32   32768   (attn tile)
//   sOm  @97280  [128][256]f32  131072   (omega, pre-softmax)
//   sV   @228352 [64][264] bf16  33792   (B^T of v: [d][k2])
//   sOB  @0      [128][264]bf16  67584   (post-softmax omega; aliases dead Wd/KT/Q/At)
//   TDM raw wd f32 staged @61440 (50176B), consumed before sQ/sAt/... are written.
// ---------------------------------------------------------------------------
__global__ __launch_bounds__(256) void attn_fused_kernel(const bf16_t* __restrict__ qbuf,
                                                         const bf16_t* __restrict__ kbuf,
                                                         const bf16_t* __restrict__ vbuf,
                                                         const float* __restrict__ wd,
                                                         bf16_t* __restrict__ outbuf) {
  extern __shared__ __align__(16) char smem[];
  bf16_t* sWd = (bf16_t*)(smem + 0);
  bf16_t* sKT = (bf16_t*)(smem + 36864);
  bf16_t* sQ  = (bf16_t*)(smem + 46080);
  bf16_t* sSp = sQ;                                  // sparse A aliases q
  float*  sAt = (float*) (smem + 64512);
  float*  sOm = (float*) (smem + 97280);
  bf16_t* sV  = (bf16_t*)(smem + 228352);
  bf16_t* sOB = (bf16_t*)(smem + 0);
  float*  sWdRaw = (float*)(smem + 61440);           // TDM staging

  const int t = threadIdx.x, lane = t & 31, w = t >> 5;
  const int nBase = blockIdx.x * 128;
  const int g = blockIdx.y, b = blockIdx.z;
  const int bg = b * NG + g;

  // --- Phase 0: TDM DMA of wd (256x49 f32, 50176B) into LDS staging -------
  if (t < 32) {
    unsigned int ldsAddr = (unsigned int)(size_t)(void*)sWdRaw;
    unsigned long long ga = (unsigned long long)(size_t)wd;
    u32x4 g0;
    g0[0] = 1u;                                       // count=1, user mode
    g0[1] = ldsAddr;                                  // lds_addr [63:32]
    g0[2] = (unsigned int)(ga & 0xFFFFFFFFull);       // global_addr lo
    g0[3] = (unsigned int)((ga >> 32) & 0x01FFFFFFull) | (2u << 30); // type=2
    i32x8 g1;
    g1[0] = (int)(2u << 16);                          // data_size=4B
    g1[1] = (int)((unsigned)(K2 * S2) << 16);         // tensor_dim0[15:0]=12544
    g1[2] = (int)(1u << 16);                          // tensor_dim0 hi=0 | tensor_dim1=1
    g1[3] = (int)((unsigned)(K2 * S2) << 16);         // tile_dim0=12544
    g1[4] = 1;                                        // tile_dim1=1
    g1[5] = (int)(unsigned)(K2 * S2);                 // tensor_dim0_stride
    g1[6] = 0;
    g1[7] = 0;
    i32x4 gz; gz[0] = gz[1] = gz[2] = gz[3] = 0;
    i32x8 gz8;
#pragma unroll
    for (int i = 0; i < 8; ++i) gz8[i] = 0;
    // 6-arg form (clang-23 / therock-10.0 headers): (g0, g1, g2, g3, g4, cpol)
    __builtin_amdgcn_tensor_load_to_lds(g0, g1, gz, gz, gz8, 0);
    __builtin_amdgcn_s_wait_tensorcnt(0);             // TENSORcnt==0 -> data in LDS
  }
  __syncthreads();

  // --- Phase 1: convert wd f32 -> bf16 B^T [k2][s pad 64] ------------------
  {
    const float* src = sWdRaw + t * S2;               // row t of wd
    bf16_t* dst = sWd + t * 72;
#pragma unroll
    for (int s = 0; s < 64; ++s) dst[s] = (s < S2) ? (bf16_t)src[s] : (bf16_t)0.0f;
  }
  __syncthreads();

  // --- Phase 2: stage q, kT, v (overwrites TDM staging) --------------------
  { // sQ[n][d]
    int r = t >> 1, c0 = (t & 1) * 32;
    const bf16_t* src = qbuf + ((size_t)bg * HW + nBase + r) * 64 + c0;
#pragma unroll
    for (int j = 0; j < 32; ++j) sQ[r * 72 + c0 + j] = src[j];
  }
  { // sKT[s][d]
    int s = t >> 2, c0 = (t & 3) * 16;
    const bf16_t* src = kbuf + ((size_t)bg * 64 + s) * 64 + c0;
#pragma unroll
    for (int j = 0; j < 16; ++j) sKT[s * 72 + c0 + j] = src[j];
  }
  { // sV[d][k2]
    int d = t >> 2, c0 = (t & 3) * 64;
    const bf16_t* src = vbuf + ((size_t)bg * 64 + d) * 256 + c0;
#pragma unroll
    for (int j = 0; j < 64; ++j) sV[d * 264 + c0 + j] = src[j];
  }
  __syncthreads();

  const int rowAdd = (lane >> 4) * 8, col = lane & 15;

  // --- Phase 3: attn = q.kT * scale  (M=128, N=64, K=64) -------------------
  {
    v8f acc[4];
#pragma unroll
    for (int i = 0; i < 4; ++i) acc[i] = zero_v8f();
#pragma unroll
    for (int kc = 0; kc < 2; ++kc) {
      v16bf af = load_a_frag(sQ + (w * 16) * 72 + kc * 32, 72, lane);
#pragma unroll
      for (int nt = 0; nt < 4; ++nt) {
        v16bf bfr = load_b_frag(sKT + (nt * 16) * 72 + kc * 32, 72, lane);
        WMMA_BF16(acc[nt], af, bfr);
      }
    }
#pragma unroll
    for (int nt = 0; nt < 4; ++nt)
#pragma unroll
      for (int r = 0; r < 8; ++r)
        sAt[(w * 16 + r + rowAdd) * 64 + nt * 16 + col] = acc[nt][r] * 0.125f; // hd^-0.5
  }
  __syncthreads();

  // --- Phase 4: per-row top-32-of-49, write sparse bf16 A ------------------
  if (t < 128) {
    const float* row = sAt + t * 64;
    bf16_t* srow = sSp + t * 72;
#pragma unroll 1
    for (int s = 0; s < 64; ++s) {
      float v = 0.0f;
      if (s < S2) {
        float a = row[s];
        int cnt = 0;
        for (int j = 0; j < S2; ++j) cnt += (row[j] > a);
        if (cnt < TOPK) v = a;                        // rank < 32 -> keep
      }
      srow[s] = (bf16_t)v;
    }
  }
  __syncthreads();

  // --- Phase 5: omega = sparse . wd^T  (M=128, N=256, K=64) ----------------
  {
    v8f oacc[16];
#pragma unroll
    for (int i = 0; i < 16; ++i) oacc[i] = zero_v8f();
#pragma unroll
    for (int kc = 0; kc < 2; ++kc) {
      v16bf af = load_a_frag(sSp + (w * 16) * 72 + kc * 32, 72, lane);
#pragma unroll
      for (int nt = 0; nt < 16; ++nt) {
        v16bf bfr = load_b_frag(sWd + (nt * 16) * 72 + kc * 32, 72, lane);
        WMMA_BF16(oacc[nt], af, bfr);
      }
    }
#pragma unroll
    for (int nt = 0; nt < 16; ++nt)
#pragma unroll
      for (int r = 0; r < 8; ++r)
        sOm[(w * 16 + r + rowAdd) * 256 + nt * 16 + col] = oacc[nt][r];
  }
  __syncthreads();

  // --- Phase 6: softmax over K2=256 per row -> bf16 A (sOB) ----------------
  if (t < 128) {
    const float* row = sOm + t * 256;
    float m = -3.0e38f;
    for (int j = 0; j < K2; ++j) m = fmaxf(m, row[j]);
    float sum = 0.0f;
    for (int j = 0; j < K2; ++j) sum += __expf(row[j] - m);
    float inv = 1.0f / sum;
    bf16_t* orow = sOB + t * 264;
    for (int j = 0; j < K2; ++j) orow[j] = (bf16_t)(__expf(row[j] - m) * inv);
  }
  __syncthreads();

  // --- Phase 7: out = omega . v  (M=128, N=64, K=256) ----------------------
  {
    v8f vacc[4];
#pragma unroll
    for (int i = 0; i < 4; ++i) vacc[i] = zero_v8f();
#pragma unroll
    for (int kc = 0; kc < 8; ++kc) {
      v16bf af = load_a_frag(sOB + (w * 16) * 264 + kc * 32, 264, lane);
#pragma unroll
      for (int nt = 0; nt < 4; ++nt) {
        v16bf bfr = load_b_frag(sV + (nt * 16) * 264 + kc * 32, 264, lane);
        WMMA_BF16(vacc[nt], af, bfr);
      }
    }
#pragma unroll
    for (int nt = 0; nt < 4; ++nt)
#pragma unroll
      for (int r = 0; r < 8; ++r) {
        int n = nBase + w * 16 + r + rowAdd;
        int d = nt * 16 + col;
        outbuf[((size_t)bg * HW + n) * 64 + d] = (bf16_t)vacc[nt][r];
      }
  }
}

// ---------------------------------------------------------------------------
// proj: y = proj_w(768x768) * out(768xHW) + bias + x   (f32 output)
// ---------------------------------------------------------------------------
__global__ __launch_bounds__(256) void proj_kernel(const bf16_t* __restrict__ outbuf,
                                                   const float* __restrict__ proj_w,
                                                   const float* __restrict__ proj_b,
                                                   const float* __restrict__ x,
                                                   float* __restrict__ y) {
  const int nBase = blockIdx.x * 128;
  const int mBase = blockIdx.y * 128;
  const int b = blockIdx.z;
  __shared__ bf16_t sA[128 * 40];
  __shared__ bf16_t sB[128 * 40];
  const int t = threadIdx.x, lane = t & 31, w = t >> 5;
  v8f acc[8];
#pragma unroll
  for (int i = 0; i < 8; ++i) acc[i] = zero_v8f();

  for (int k0 = 0; k0 < CIN; k0 += 32) {
    { // A: proj_w rows
      int m = t >> 1, c0 = (t & 1) * 16;
      const float* src = proj_w + (size_t)(mBase + m) * CIN + k0 + c0;
      bf16_t* dst = sA + m * 40 + c0;
#pragma unroll
      for (int j = 0; j < 16; ++j) dst[j] = (bf16_t)src[j];
    }
    { // B^T: sB[n][kk] = outbuf[b][g][nBase+n][d0+kk] ; chunk stays within one g
      int gk = k0 >> 6, d0 = k0 & 63;
      int n = t >> 1, kk0 = (t & 1) * 16;
      const bf16_t* src = outbuf + (((size_t)b * NG + gk) * HW + nBase + n) * 64 + d0 + kk0;
      bf16_t* dst = sB + n * 40 + kk0;
#pragma unroll
      for (int j = 0; j < 16; ++j) dst[j] = src[j];
    }
    __syncthreads();
    if (k0 + 32 < CIN)
      __builtin_prefetch(proj_w + (size_t)(mBase + (t >> 1)) * CIN + k0 + 32, 0, 1);
    v16bf af = load_a_frag(sA + (w * 16) * 40, 40, lane);
#pragma unroll
    for (int nt = 0; nt < 8; ++nt) {
      v16bf bfr = load_b_frag(sB + (nt * 16) * 40, 40, lane);
      WMMA_BF16(acc[nt], af, bfr);
    }
    __syncthreads();
  }
  const int rowAdd = (lane >> 4) * 8, col = lane & 15;
#pragma unroll
  for (int nt = 0; nt < 8; ++nt)
#pragma unroll
    for (int r = 0; r < 8; ++r) {
      int o = mBase + w * 16 + r + rowAdd;
      int n = nBase + nt * 16 + col;
      size_t idx = ((size_t)b * CIN + o) * HW + n;
      y[idx] = acc[nt][r] + proj_b[o] + x[idx];       // bias + residual
    }
}

// ---------------------------------------------------------------------------
extern "C" void kernel_launch(void* const* d_in, const int* in_sizes, int n_in,
                              void* d_out, int out_size, void* d_ws, size_t ws_size,
                              hipStream_t stream) {
  const float* x      = (const float*)d_in[0];
  const float* ctx    = (const float*)d_in[1];
  const float* wq     = (const float*)d_in[2];
  const float* wk     = (const float*)d_in[3];
  const float* wd     = (const float*)d_in[4];
  const float* proj_w = (const float*)d_in[5];
  const float* proj_b = (const float*)d_in[6];
  float* y = (float*)d_out;

  // Workspace carve-up (~78.2MB)
  char* ws = (char*)d_ws;
  bf16_t* qbuf   = (bf16_t*)(ws);                       // 38,535,168 B
  bf16_t* outbuf = (bf16_t*)(ws + 38535168);            // 38,535,168 B
  bf16_t* kbuf   = (bf16_t*)(ws + 77070336);            //    196,608 B
  bf16_t* vbuf   = (bf16_t*)(ws + 77266944);            //    786,432 B
  float*  ctxp   = (float*) (ws + 78053376);            //    150,528 B

  pool_ctx_kernel<<<147, 256, 0, stream>>>(ctx, ctxp);
  pool_v_kernel<<<1536, 256, 0, stream>>>(x, vbuf);
  kproj_kernel<<<dim3(6, 2), 256, 0, stream>>>(wk, ctxp, kbuf);
  qproj_kernel<<<dim3(98, 6, 2), 256, 0, stream>>>(x, wq, qbuf);
  attn_fused_kernel<<<dim3(98, 12, 2), 256, 262144, stream>>>(qbuf, kbuf, vbuf, wd, outbuf);
  proj_kernel<<<dim3(98, 6, 2), 256, 0, stream>>>(outbuf, proj_w, proj_b, x, y);
}
